// Layer_Block_39444979646721
// MI455X (gfx1250) — matrix operand order repaired and checked by reading, not compiled, not run on hardware
//
#include <hip/hip_runtime.h>
#include <hip/hip_bf16.h>
#include <math.h>

#define DIM    768
#define NHEADS 12
#define HDIM   64
#define FFNDIM 3072
#define SEQ    4096
#define BATCH  2
#define ROWS   (BATCH * SEQ)   // 8192

typedef __attribute__((ext_vector_type(16))) __bf16        v16bf;
typedef __attribute__((ext_vector_type(8)))  float         v8f;
typedef __attribute__((ext_vector_type(4)))  unsigned int  u32x4;
typedef __attribute__((ext_vector_type(8)))  int           i32x8;
typedef __attribute__((ext_vector_type(4)))  int           i32x4;

union FragU {
  v16bf bf;
  u32x4 q[2];
  unsigned short h[16];
};
union U8 {
  u32x4 v;
  unsigned short h[8];
};

__device__ __forceinline__ unsigned short f2bf(float f) {
  unsigned int u = __float_as_uint(f);
  u += 0x7FFFu + ((u >> 16) & 1u);           // round-to-nearest-even
  return (unsigned short)(u >> 16);
}
__device__ __forceinline__ float bf2f(unsigned short h) {
  return __uint_as_float(((unsigned int)h) << 16);
}

// A fragment (16x32 bf16, ISA 7.12.2): lane holds row (lane&15); two
// contiguous 8-element chunks at k0 and k0+16, k0 = (lane<16 ? 0 : 8) + base.
__device__ __forceinline__ v16bf load_a(const unsigned short* row, int k0) {
  FragU f;
  f.q[0] = *(const u32x4*)(row + k0);
  f.q[1] = *(const u32x4*)(row + k0 + 16);
  return f.bf;
}
// B fragment (32x16 bf16): lane holds column (lane&15); two contiguous
// 8-element chunks at k0 and k0+16, k0 = (lane<16 ? 0 : 8) + base, from a [N][K] tile.
__device__ __forceinline__ v16bf load_b(const unsigned short* col, int k0) {
  FragU f;
  f.q[0] = *(const u32x4*)(col + k0);
  f.q[1] = *(const u32x4*)(col + k0 + 16);
  return f.bf;
}

__device__ __forceinline__ v8f wmma_bf16(v16bf a, v16bf b, v8f c) {
  return __builtin_amdgcn_wmma_f32_16x16x32_bf16(false, a, false, b,
                                                 (short)0, c, false, false);
}

// branch-free tanh-GELU: 0.5u(1+tanh(z)) == u * sigmoid(2z)
__device__ __forceinline__ float gelu_f(float u) {
  float z = 0.7978845608028654f * (u + 0.044715f * u * u * u);
  return u * (1.0f / (1.0f + __expf(-2.0f * z)));
}

// ---------------------------------------------------------------- LayerNorm
__global__ __launch_bounds__(256) void ln_bf16_kernel(
    const float* __restrict__ x, const float* __restrict__ g,
    const float* __restrict__ b, unsigned short* __restrict__ out) {
  __shared__ float red[256];
  const int row = blockIdx.x;
  const int t = threadIdx.x;
  const float* xr = x + (size_t)row * DIM;
  float v0 = xr[t], v1 = xr[t + 256], v2 = xr[t + 512];
  red[t] = v0 + v1 + v2;
  __syncthreads();
  for (int o = 128; o > 0; o >>= 1) { if (t < o) red[t] += red[t + o]; __syncthreads(); }
  float mu = red[0] * (1.0f / DIM);
  __syncthreads();
  float d0 = v0 - mu, d1 = v1 - mu, d2 = v2 - mu;
  red[t] = d0 * d0 + d1 * d1 + d2 * d2;
  __syncthreads();
  for (int o = 128; o > 0; o >>= 1) { if (t < o) red[t] += red[t + o]; __syncthreads(); }
  float rstd = rsqrtf(red[0] * (1.0f / DIM) + 1e-5f);
  unsigned short* orow = out + (size_t)row * DIM;
  orow[t]       = f2bf(d0 * rstd * g[t]       + b[t]);
  orow[t + 256] = f2bf(d1 * rstd * g[t + 256] + b[t + 256]);
  orow[t + 512] = f2bf(d2 * rstd * g[t + 512] + b[t + 512]);
}

// ------------------------------------------------------------ f32 -> bf16
__global__ void cvt_bf16_kernel(const float* __restrict__ in,
                                unsigned short* __restrict__ out, int n) {
  int i = blockIdx.x * blockDim.x + threadIdx.x;
  if (i < n) out[i] = f2bf(in[i]);
}

// ------------------------------------------------------- WMMA bf16 GEMM
// C[M,N] = A[M,K] @ B[K,N] + bias, epilogue by compile-time MODE.
// Block tile 64x256, 8 waves, wave tile 32x64 (8 WMMAs / 32-K step).
// A tile staged by the Tensor Data Mover (one TDM descriptor per K step,
// issued by wave 0, LDS rows padded 32->40 bf16 via D# pad fields).
#define GMODE_BF16      0
#define GMODE_BF16_GELU 1
#define GMODE_F32_RES   2

template <int MODE>
__global__ __launch_bounds__(256) void gemm_bf16_kernel(
    const unsigned short* __restrict__ A,   // [M,K] bf16
    const unsigned short* __restrict__ Bw,  // [K,N] bf16
    const float* __restrict__ bias,         // [N]
    void* __restrict__ Cout,
    const float* __restrict__ Res,          // [M,N] f32 (mode 2)
    int M, int N, int K) {
  __shared__ alignas(16) unsigned short As[64][40];   // TDM dest (padded)
  __shared__ alignas(16) unsigned short Bs[256][40];  // transposed [n][k]

  const int t = threadIdx.x;
  const int lane = t & 31;
  const int w = t >> 5;
  const int wm = w & 1, wn = w >> 1;          // 2 x 4 wave grid
  const int m0 = blockIdx.x * 64;
  const int n0 = blockIdx.y * 256;

  const int cl   = lane & 15;
  const int aRow = wm * 32 + cl;
  const int aK0  = (lane < 16) ? 0 : 8;
  const int bCol = wn * 64 + cl;
  const int bK0  = (lane < 16) ? 0 : 8;

  const v8f zero8 = {0.f, 0.f, 0.f, 0.f, 0.f, 0.f, 0.f, 0.f};
  v8f acc[2][4];
  #pragma unroll
  for (int mi = 0; mi < 2; ++mi)
    #pragma unroll
    for (int ni = 0; ni < 4; ++ni) acc[mi][ni] = zero8;

  // invariant pieces of the TDM descriptor (ISA 8.3/8.4)
  const unsigned int ldsoff = (unsigned int)(size_t)&As[0][0];
  // group1 d0: data_size=2B (code 1), pad_enable, pad_interval=16DW (code 3),
  // pad_amount=4DW (code 3)  -> LDS row stride 40 bf16
  const unsigned int g1d0 = (1u << 16) | (1u << 20) | (3u << 22) | (3u << 25);

  for (int k0 = 0; k0 < K; k0 += 32) {
    __syncthreads();
    if (w == 0) {
      // ---- TDM: load A tile [64 rows][32 cols] -> LDS (wave 0 issues) ----
      unsigned long long ga =
          (unsigned long long)(size_t)A + ((size_t)m0 * K + k0) * 2ull;
      u32x4 g0;
      g0[0] = 1u;                              // count=1, user descriptor
      g0[1] = ldsoff;                          // lds_addr
      g0[2] = (unsigned int)(ga & 0xFFFFFFFFull);
      g0[3] = (unsigned int)((ga >> 32) & 0x1FFFFFFull) | (2u << 30); // type=2
      i32x8 g1;
      g1[0] = (int)g1d0;
      g1[1] = (int)(((unsigned int)K & 0xFFFFu) << 16);        // dim0 lo
      g1[2] = (int)((((unsigned int)K >> 16) & 0xFFFFu) |
                    (((unsigned int)M & 0xFFFFu) << 16));      // dim0 hi|dim1 lo
      g1[3] = (int)(((((unsigned int)M >> 16) & 0xFFFFu)) |
                    (32u << 16));                              // dim1 hi|tile0=32
      g1[4] = 64;                                              // tile1=64
      g1[5] = (int)(unsigned int)K;                            // stride0 lo
      g1[6] = 0;                                               // stride0 hi
      g1[7] = 0;
      i32x4 gz4 = {0, 0, 0, 0};
      i32x8 gz8 = {0, 0, 0, 0, 0, 0, 0, 0};
      __builtin_amdgcn_tensor_load_to_lds(g0, g1, gz4, gz4, gz8, 0);
    }
    // ---- all waves: B tile 32x256 loaded transposed into Bs[n][kk] ----
    #pragma unroll
    for (int i = 0; i < 4; ++i) {
      int chunk = t + i * 256;                  // 0..1023 (8 bf16 each)
      int kk = chunk >> 5;
      int n8 = (chunk & 31) * 8;
      U8 tmp;
      tmp.v = *(const u32x4*)&Bw[(size_t)(k0 + kk) * N + n0 + n8];
      #pragma unroll
      for (int j = 0; j < 8; ++j) Bs[n8 + j][kk] = tmp.h[j];
    }
    if (w == 0) __builtin_amdgcn_s_wait_tensorcnt(0);
    __syncthreads();

    v16bf a0 = load_a(&As[aRow][0],      aK0);
    v16bf a1 = load_a(&As[aRow + 16][0], aK0);
    #pragma unroll
    for (int ni = 0; ni < 4; ++ni) {
      v16bf b = load_b(&Bs[bCol + ni * 16][0], bK0);
      acc[0][ni] = wmma_bf16(a0, b, acc[0][ni]);
      acc[1][ni] = wmma_bf16(a1, b, acc[1][ni]);
    }
  }

  const int rHalf = (lane >> 4) * 8;
  #pragma unroll
  for (int mi = 0; mi < 2; ++mi)
    #pragma unroll
    for (int ni = 0; ni < 4; ++ni)
      #pragma unroll
      for (int e = 0; e < 8; ++e) {
        int gr = m0 + wm * 32 + mi * 16 + e + rHalf;
        int gc = n0 + wn * 64 + ni * 16 + cl;
        float val = acc[mi][ni][e] + bias[gc];
        size_t oi = (size_t)gr * N + gc;
        if (MODE == GMODE_F32_RES) {
          ((float*)Cout)[oi] = Res[oi] + val;
        } else if (MODE == GMODE_BF16_GELU) {
          ((unsigned short*)Cout)[oi] = f2bf(gelu_f(val));
        } else {
          ((unsigned short*)Cout)[oi] = f2bf(val);
        }
      }
}

// ------------------------------------------------ Flash attention (causal)
// One block per (batch, head, 64-query tile); 4 waves x 16 query rows.
__global__ __launch_bounds__(128) void flash_attn_kernel(
    const unsigned short* __restrict__ Q,
    const unsigned short* __restrict__ Kc,
    const unsigned short* __restrict__ Vc,
    unsigned short* __restrict__ ctx) {
  __shared__ alignas(16) unsigned short Qs[64][72];    // [q][dh], pre-scaled
  __shared__ alignas(16) unsigned short Ks[32][72];    // [key][dh]
  __shared__ alignas(16) unsigned short Vt[64][40];    // [dh][key] (transposed)
  __shared__ alignas(16) unsigned short Ps[4][16][40]; // per-wave P tile

  const int t = threadIdx.x;
  const int lane = t & 31;
  const int w = t >> 5;
  const int qbase = blockIdx.x * 64;
  const int b = blockIdx.y / NHEADS;
  const int h = blockIdx.y % NHEADS;
  const size_t rowBase = (size_t)b * SEQ;
  const int hoff = h * HDIM;

  // stage Q, folding the 1/sqrt(Dh) score scale into Q
  for (int i = 0; i < 32; ++i) {
    int linear = t + i * 128;
    int r = linear >> 6, d = linear & 63;
    float qv = bf2f(Q[(rowBase + qbase + r) * DIM + hoff + d]) * 0.125f;
    Qs[r][d] = f2bf(qv);
  }

  const int cl = lane & 15;
  const int rHalf = (lane >> 4) * 8;
  const int aK0 = (lane < 16) ? 0 : 8;
  const int bK0 = (lane < 16) ? 0 : 8;

  const v8f zero8 = {0.f, 0.f, 0.f, 0.f, 0.f, 0.f, 0.f, 0.f};
  float m[8], l[8];
  v8f o[4];
  #pragma unroll
  for (int e = 0; e < 8; ++e) { m[e] = -1e30f; l[e] = 0.0f; }
  o[0] = zero8; o[1] = zero8; o[2] = zero8; o[3] = zero8;

  for (int kt = 0; kt < qbase + 64; kt += 32) {   // causal key range (uniform)
    __syncthreads();
    // stage K [32][64] vectorized, V transposed into Vt[dh][key]
    #pragma unroll
    for (int i = 0; i < 2; ++i) {
      int chunk = t + i * 128;                    // 0..255
      int r = chunk >> 3, d8 = (chunk & 7) * 8;
      *(u32x4*)&Ks[r][d8] =
          *(const u32x4*)&Kc[(rowBase + kt + r) * DIM + hoff + d8];
    }
    #pragma unroll
    for (int i = 0; i < 16; ++i) {
      int linear = t + i * 128;
      int r = linear >> 6, d = linear & 63;
      Vt[d][r] = Vc[(rowBase + kt + r) * DIM + hoff + d];
    }
    __syncthreads();

    // scores S[16q][32k] = (Q*scale) @ K^T  (K tile is already B-transposed)
    v8f s0 = zero8, s1 = zero8;
    const unsigned short* qrow = &Qs[w * 16 + cl][0];
    #pragma unroll
    for (int dho = 0; dho < 64; dho += 32) {
      v16bf aq  = load_a(qrow, aK0 + dho);
      v16bf bk0 = load_b(&Ks[cl][0],      bK0 + dho);
      v16bf bk1 = load_b(&Ks[16 + cl][0], bK0 + dho);
      s0 = wmma_bf16(aq, bk0, s0);
      s1 = wmma_bf16(aq, bk1, s1);
    }

    // online softmax over the 32-key tile (per-row, 16-lane butterflies)
    #pragma unroll
    for (int e = 0; e < 8; ++e) {
      int qrowg = qbase + w * 16 + e + rHalf;
      float x0 = (kt + cl      <= qrowg) ? s0[e] : -1e30f;
      float x1 = (kt + 16 + cl <= qrowg) ? s1[e] : -1e30f;
      float mx = fmaxf(x0, x1);
      mx = fmaxf(mx, __shfl_xor(mx, 1, 32));
      mx = fmaxf(mx, __shfl_xor(mx, 2, 32));
      mx = fmaxf(mx, __shfl_xor(mx, 4, 32));
      mx = fmaxf(mx, __shfl_xor(mx, 8, 32));
      float mnew = fmaxf(m[e], mx);
      float p0 = __expf(x0 - mnew);
      float p1 = __expf(x1 - mnew);
      float se = p0 + p1;
      se += __shfl_xor(se, 1, 32);
      se += __shfl_xor(se, 2, 32);
      se += __shfl_xor(se, 4, 32);
      se += __shfl_xor(se, 8, 32);
      float sc = __expf(m[e] - mnew);
      l[e] = l[e] * sc + se;
      m[e] = mnew;
      #pragma unroll
      for (int dj = 0; dj < 4; ++dj) o[dj][e] *= sc;
      Ps[w][e + rHalf][cl]      = f2bf(p0);   // C-layout -> A-layout via LDS
      Ps[w][e + rHalf][16 + cl] = f2bf(p1);
    }
    __syncthreads();

    // O[16q][64d] += P[16q][32k] @ V[32k][64d]   (Vt is B-transposed)
    v16bf pa = load_a(&Ps[w][cl][0], aK0);
    #pragma unroll
    for (int dj = 0; dj < 4; ++dj) {
      v16bf vb = load_b(&Vt[dj * 16 + cl][0], bK0);
      o[dj] = wmma_bf16(pa, vb, o[dj]);
    }
  }

  // finalize: divide by softmax sum, emit bf16 ctx for the Wo GEMM
  #pragma unroll
  for (int dj = 0; dj < 4; ++dj)
    #pragma unroll
    for (int e = 0; e < 8; ++e) {
      size_t oi = (rowBase + qbase + w * 16 + e + rHalf) * DIM + hoff +
                  dj * 16 + cl;
      ctx[oi] = f2bf(o[dj][e] * (1.0f / l[e]));
    }
}

// ----------------------------------------------------------------- driver
extern "C" void kernel_launch(void* const* d_in, const int* in_sizes, int n_in,
                              void* d_out, int out_size, void* d_ws,
                              size_t ws_size, hipStream_t stream) {
  (void)in_sizes; (void)n_in; (void)out_size; (void)ws_size;

  const float* x    = (const float*)d_in[0];
  const float* ln1g = (const float*)d_in[1];
  const float* ln1b = (const float*)d_in[2];
  const float* Wq   = (const float*)d_in[3];
  const float* bq   = (const float*)d_in[4];
  const float* Wk   = (const float*)d_in[5];
  const float* bk   = (const float*)d_in[6];
  const float* Wv   = (const float*)d_in[7];
  const float* bv   = (const float*)d_in[8];
  const float* Wo   = (const float*)d_in[9];
  const float* bo   = (const float*)d_in[10];
  const float* ln2g = (const float*)d_in[11];
  const float* ln2b = (const float*)d_in[12];
  const float* W1   = (const float*)d_in[13];
  const float* b1   = (const float*)d_in[14];
  const float* W2   = (const float*)d_in[15];
  const float* b2   = (const float*)d_in[16];

  char* ws = (char*)d_ws;
  size_t off = 0;
  auto alloc = [&](size_t bytes) -> char* {
    char* p = ws + off;
    off += (bytes + 255) & ~(size_t)255;
    return p;
  };

  unsigned short* hbf  = (unsigned short*)alloc((size_t)ROWS * DIM * 2);
  unsigned short* qbf  = (unsigned short*)alloc((size_t)ROWS * DIM * 2);
  unsigned short* kbf  = (unsigned short*)alloc((size_t)ROWS * DIM * 2);
  unsigned short* vbf  = (unsigned short*)alloc((size_t)ROWS * DIM * 2);
  unsigned short* ctxb = (unsigned short*)alloc((size_t)ROWS * DIM * 2);
  float*          x1   = (float*)        alloc((size_t)ROWS * DIM * 4);
  unsigned short* a1bf = (unsigned short*)alloc((size_t)ROWS * FFNDIM * 2);
  unsigned short* Wqb  = (unsigned short*)alloc((size_t)DIM * DIM * 2);
  unsigned short* Wkb  = (unsigned short*)alloc((size_t)DIM * DIM * 2);
  unsigned short* Wvb  = (unsigned short*)alloc((size_t)DIM * DIM * 2);
  unsigned short* Wob  = (unsigned short*)alloc((size_t)DIM * DIM * 2);
  unsigned short* W1b  = (unsigned short*)alloc((size_t)DIM * FFNDIM * 2);
  unsigned short* W2b  = (unsigned short*)alloc((size_t)FFNDIM * DIM * 2);

  // weight downconvert (bf16 WMMA path)
  const int nDD = DIM * DIM, nDF = DIM * FFNDIM;
  cvt_bf16_kernel<<<(nDD + 255) / 256, 256, 0, stream>>>(Wq, Wqb, nDD);
  cvt_bf16_kernel<<<(nDD + 255) / 256, 256, 0, stream>>>(Wk, Wkb, nDD);
  cvt_bf16_kernel<<<(nDD + 255) / 256, 256, 0, stream>>>(Wv, Wvb, nDD);
  cvt_bf16_kernel<<<(nDD + 255) / 256, 256, 0, stream>>>(Wo, Wob, nDD);
  cvt_bf16_kernel<<<(nDF + 255) / 256, 256, 0, stream>>>(W1, W1b, nDF);
  cvt_bf16_kernel<<<(nDF + 255) / 256, 256, 0, stream>>>(W2, W2b, nDF);

  // h = LN1(x)  (fused f32 -> bf16)
  ln_bf16_kernel<<<ROWS, 256, 0, stream>>>(x, ln1g, ln1b, hbf);

  // q, k, v projections  (block tile 64 x 256)
  dim3 gDD(ROWS / 64, DIM / 256);
  gemm_bf16_kernel<GMODE_BF16><<<gDD, 256, 0, stream>>>(
      hbf, Wqb, bq, qbf, nullptr, ROWS, DIM, DIM);
  gemm_bf16_kernel<GMODE_BF16><<<gDD, 256, 0, stream>>>(
      hbf, Wkb, bk, kbf, nullptr, ROWS, DIM, DIM);
  gemm_bf16_kernel<GMODE_BF16><<<gDD, 256, 0, stream>>>(
      hbf, Wvb, bv, vbf, nullptr, ROWS, DIM, DIM);

  // causal flash attention -> ctx (bf16)
  dim3 gFA(SEQ / 64, BATCH * NHEADS);
  flash_attn_kernel<<<gFA, 128, 0, stream>>>(qbf, kbf, vbf, ctxb);

  // x1 = x + ctx @ Wo + bo
  gemm_bf16_kernel<GMODE_F32_RES><<<gDD, 256, 0, stream>>>(
      ctxb, Wob, bo, x1, x, ROWS, DIM, DIM);

  // h = LN2(x1)
  ln_bf16_kernel<<<ROWS, 256, 0, stream>>>(x1, ln2g, ln2b, hbf);

  // a1 = gelu(h @ W1 + b1)
  dim3 gDF(ROWS / 64, FFNDIM / 256);
  gemm_bf16_kernel<GMODE_BF16_GELU><<<gDF, 256, 0, stream>>>(
      hbf, W1b, b1, a1bf, nullptr, ROWS, FFNDIM, DIM);

  // out = x1 + a1 @ W2 + b2
  gemm_bf16_kernel<GMODE_F32_RES><<<gDD, 256, 0, stream>>>(
      a1bf, W2b, b2, (float*)d_out, x1, ROWS, DIM, FFNDIM);
}
